// MHA_47132971107054
// MI455X (gfx1250) — matrix operand order repaired and checked by reading, not compiled, not run on hardware
//
#include <hip/hip_runtime.h>

#define HEADS 8
#define DIM   64
#define BATCH 4
#define SEQ   2048
#define BM    16
#define SS    2056   // padded S row stride in halves (mult of 8 for b128 alignment)

typedef __attribute__((ext_vector_type(16))) _Float16 v16h;
typedef __attribute__((ext_vector_type(8)))  _Float16 v8h;
typedef __attribute__((ext_vector_type(4)))  _Float16 v4h;
typedef __attribute__((ext_vector_type(8)))  float    v8f;
typedef __attribute__((ext_vector_type(4)))  float    v4f;

union V16U { v16h v; v8h h[2]; };

__global__ __launch_bounds__(256) void mha_fused(const float* __restrict__ x,
                                                 float* __restrict__ out_o,
                                                 float* __restrict__ out_qk)
{
  __shared__ alignas(16) _Float16 S[BM * SS];        // 65792 B : S then P (f16)
  __shared__ alignas(16) _Float16 Kt[8][16][72];     // 18432 B : per-wave K tile [l][d]
  __shared__ alignas(16) _Float16 Vt[8][64][40];     // 40960 B : per-wave V chunk [d][jl]
  __shared__ alignas(16) _Float16 Qt[16][72];        //  2304 B : Q tile [l][d]
  __shared__ float Osm[BM * DIM];                    //  4096 B
  __shared__ float redmax[16][16];
  __shared__ float redsum[16][16];
  __shared__ float rowmax[16];
  __shared__ float rowinv[16];

  const int tid  = threadIdx.x;
  const int wave = tid >> 5;
  const int lane = tid & 31;
  const int lm   = lane & 15;   // row (A) or col (B/C) index within tile
  const int g    = lane >> 4;   // lane half-group

  const int b  = blockIdx.z;
  const int h  = blockIdx.y;
  const int i0 = blockIdx.x * BM;

  // x layout: (b, C, L) with C = [3][H][D], L contiguous
  const float* qbase = x + (size_t)((b * 3 * HEADS) + h) * DIM * SEQ;
  const float* kbase = x + (size_t)((b * 3 * HEADS) + HEADS + h) * DIM * SEQ;
  const float* vbase = x + (size_t)((b * 3 * HEADS) + 2 * HEADS + h) * DIM * SEQ;

  // ---- stage Q tile (16 rows x 64 dims) as f16; zero O accumulator ----
  {
    int d  = tid >> 2;
    int l0 = (tid & 3) * 4;
    float4 f = *(const float4*)&qbase[(size_t)d * SEQ + i0 + l0];
    Qt[l0 + 0][d] = (_Float16)f.x;
    Qt[l0 + 1][d] = (_Float16)f.y;
    Qt[l0 + 2][d] = (_Float16)f.z;
    Qt[l0 + 3][d] = (_Float16)f.w;
    Osm[tid] = 0.f; Osm[tid + 256] = 0.f; Osm[tid + 512] = 0.f; Osm[tid + 768] = 0.f;
  }
  __syncthreads();

  // ---- Q A-fragments (loop invariant): elem e -> K = (e&8)*2 + 8g + (e&7) ----
  V16U aQ0, aQ1;
  aQ0.h[0] = *(const v8h*)&Qt[lm][8 * g];
  aQ0.h[1] = *(const v8h*)&Qt[lm][16 + 8 * g];
  aQ1.h[0] = *(const v8h*)&Qt[lm][32 + 8 * g];
  aQ1.h[1] = *(const v8h*)&Qt[lm][48 + 8 * g];

  const float invscale = 0.35355339059327373f;  // 1/sqrt(DIM/HEADS) = 1/sqrt(8)

  // ---- Phase A: S = (Q K^T)/scale for 16x2048 row block ----
  for (int t = 0; t < 16; ++t) {
    const int j0 = (wave * 16 + t) * 16;
    // stage K tile (16 rows x 64 dims) transposed into Kt[wave][l][d]
    for (int k = 0; k < 8; ++k) {
      int d  = (lane >> 2) + 8 * k;
      int l0 = (lane & 3) * 4;
      float4 f = *(const float4*)&kbase[(size_t)d * SEQ + j0 + l0];
      Kt[wave][l0 + 0][d] = (_Float16)f.x;
      Kt[wave][l0 + 1][d] = (_Float16)f.y;
      Kt[wave][l0 + 2][d] = (_Float16)f.z;
      Kt[wave][l0 + 3][d] = (_Float16)f.w;
    }
    asm volatile("s_wait_dscnt 0" ::: "memory");

    // B-fragments: elem e -> K = (e&8)*2 + 8g + (e&7)
    V16U bK0, bK1;
    bK0.h[0] = *(const v8h*)&Kt[wave][lm][8 * g];
    bK0.h[1] = *(const v8h*)&Kt[wave][lm][16 + 8 * g];
    bK1.h[0] = *(const v8h*)&Kt[wave][lm][32 + 8 * g];
    bK1.h[1] = *(const v8h*)&Kt[wave][lm][48 + 8 * g];

    v8f acc = {};
    acc = __builtin_amdgcn_wmma_f32_16x16x32_f16(false, aQ0.v, false, bK0.v, (short)0, acc, false, false);
    acc = __builtin_amdgcn_wmma_f32_16x16x32_f16(false, aQ1.v, false, bK1.v, (short)0, acc, false, false);

    #pragma unroll
    for (int r = 0; r < 8; ++r)   // C elem r -> row (r + 8g), col lm
      S[(r + 8 * g) * SS + j0 + lm] = (_Float16)(acc[r] * invscale);
  }
  __syncthreads();

  // ---- Phase B: per-row max and exp-sum (16 threads per row) ----
  {
    const int r16 = tid >> 4, c16 = tid & 15;
    const _Float16* srow = &S[r16 * SS];
    float mx = -1e30f;
    for (int k = 0; k < 16; ++k) {
      v8h v = *(const v8h*)&srow[c16 * 8 + k * 128];
      #pragma unroll
      for (int e = 0; e < 8; ++e) mx = fmaxf(mx, (float)v[e]);
    }
    redmax[r16][c16] = mx;
    __syncthreads();
    float m = -1e30f;
    #pragma unroll
    for (int i = 0; i < 16; ++i) m = fmaxf(m, redmax[r16][i]);
    float sum = 0.f;
    for (int k = 0; k < 16; ++k) {
      v8h v = *(const v8h*)&srow[c16 * 8 + k * 128];
      #pragma unroll
      for (int e = 0; e < 8; ++e) sum += __expf((float)v[e] - m);
    }
    redsum[r16][c16] = sum;
    __syncthreads();
    float tot = 0.f;
    #pragma unroll
    for (int i = 0; i < 16; ++i) tot += redsum[r16][i];
    if (c16 == 0) { rowmax[r16] = m; rowinv[r16] = 1.0f / tot; }
  }
  __syncthreads();

  // ---- Phase C: normalize; stream P (f32, non-temporal) to qk; f16 P back to LDS ----
  {
    const size_t qkbase = ((size_t)((b * HEADS + h) * SEQ + i0)) * SEQ;
    const int j0 = tid * 8;                 // each thread owns 8 contiguous j per row
    for (int m = 0; m < 16; ++m) {
      const float mx = rowmax[m], inv = rowinv[m];
      v8h sv = *(const v8h*)&S[m * SS + j0];
      float p[8];
      #pragma unroll
      for (int e = 0; e < 8; ++e) p[e] = __expf((float)sv[e] - mx) * inv;
      v4f w0 = { p[0], p[1], p[2], p[3] };
      v4f w1 = { p[4], p[5], p[6], p[7] };
      float* dst = &out_qk[qkbase + (size_t)m * SEQ + j0];
      __builtin_nontemporal_store(w0, (v4f*)dst);        // don't evict x from L2
      __builtin_nontemporal_store(w1, (v4f*)(dst + 4));
      v8h pv;
      #pragma unroll
      for (int e = 0; e < 8; ++e) pv[e] = (_Float16)p[e];
      *(v8h*)&S[m * SS + j0] = pv;
    }
  }
  __syncthreads();

  // ---- Phase D: O += P * V ; each wave owns 256 columns of P ----
  {
    v8f o0 = {}, o1 = {}, o2 = {}, o3 = {};
    const int jw = wave * 256;
    for (int cc = 0; cc < 8; ++cc) {
      const int jb = jw + cc * 32;
      // stage V chunk (32 seq rows x 64 dims) into Vt[wave][d][jl] (natural: mem is d-major)
      for (int k = 0; k < 16; ++k) {
        int d   = (lane >> 3) + 4 * k;
        int jl0 = (lane & 7) * 4;
        float4 f = *(const float4*)&vbase[(size_t)d * SEQ + jb + jl0];
        v4h hv;
        hv[0] = (_Float16)f.x; hv[1] = (_Float16)f.y;
        hv[2] = (_Float16)f.z; hv[3] = (_Float16)f.w;
        *(v4h*)&Vt[wave][d][jl0] = hv;
      }
      asm volatile("s_wait_dscnt 0" ::: "memory");

      V16U aP;
      aP.h[0] = *(const v8h*)&S[lm * SS + jb + 16 * g];
      aP.h[1] = *(const v8h*)&S[lm * SS + jb + 16 * g + 8];

      V16U bV;
      bV.h[0] = *(const v8h*)&Vt[wave][ 0 + lm][16 * g];
      bV.h[1] = *(const v8h*)&Vt[wave][ 0 + lm][16 * g + 8];
      o0 = __builtin_amdgcn_wmma_f32_16x16x32_f16(false, aP.v, false, bV.v, (short)0, o0, false, false);
      bV.h[0] = *(const v8h*)&Vt[wave][16 + lm][16 * g];
      bV.h[1] = *(const v8h*)&Vt[wave][16 + lm][16 * g + 8];
      o1 = __builtin_amdgcn_wmma_f32_16x16x32_f16(false, aP.v, false, bV.v, (short)0, o1, false, false);
      bV.h[0] = *(const v8h*)&Vt[wave][32 + lm][16 * g];
      bV.h[1] = *(const v8h*)&Vt[wave][32 + lm][16 * g + 8];
      o2 = __builtin_amdgcn_wmma_f32_16x16x32_f16(false, aP.v, false, bV.v, (short)0, o2, false, false);
      bV.h[0] = *(const v8h*)&Vt[wave][48 + lm][16 * g];
      bV.h[1] = *(const v8h*)&Vt[wave][48 + lm][16 * g + 8];
      o3 = __builtin_amdgcn_wmma_f32_16x16x32_f16(false, aP.v, false, bV.v, (short)0, o3, false, false);
    }
    #pragma unroll
    for (int r = 0; r < 8; ++r) {
      int row = r + 8 * g;
      atomicAdd(&Osm[row * DIM +      lm], o0[r]);
      atomicAdd(&Osm[row * DIM + 16 + lm], o1[r]);
      atomicAdd(&Osm[row * DIM + 32 + lm], o2[r]);
      atomicAdd(&Osm[row * DIM + 48 + lm], o3[r]);
    }
  }
  __syncthreads();

  // ---- write O: out_o[b][h*64 + d][i0 + l] (non-temporal stream) ----
  {
    int d  = tid >> 2;
    int l0 = (tid & 3) * 4;
    v4f ov;
    ov[0] = Osm[(l0 + 0) * DIM + d];
    ov[1] = Osm[(l0 + 1) * DIM + d];
    ov[2] = Osm[(l0 + 2) * DIM + d];
    ov[3] = Osm[(l0 + 3) * DIM + d];
    float* dst = &out_o[((size_t)(b * HEADS * DIM + h * DIM + d)) * SEQ + i0 + l0];
    __builtin_nontemporal_store(ov, (v4f*)dst);
  }
}

extern "C" void kernel_launch(void* const* d_in, const int* in_sizes, int n_in,
                              void* d_out, int out_size, void* d_ws, size_t ws_size,
                              hipStream_t stream) {
  const float* x = (const float*)d_in[0];
  float* out_o  = (float*)d_out;
  float* out_qk = out_o + (size_t)BATCH * HEADS * DIM * SEQ;  // o first, then qk
  dim3 grid(SEQ / BM, HEADS, BATCH);
  mha_fused<<<grid, 256, 0, stream>>>(x, out_o, out_qk);
  (void)in_sizes; (void)n_in; (void)out_size; (void)d_ws; (void)ws_size;
}
